// CounterfactualURHGNModel_41918880809423
// MI455X (gfx1250) — matrix operand order repaired and checked by reading, not compiled, not run on hardware
//
#include <hip/hip_runtime.h>
#include <math.h>

// ---------------------------------------------------------------------------
// MI455X (gfx1250, wave32) implementation of the URH-GNN pipeline.
// Dense GEMMs: V_WMMA_F32_16X16X32_F16 with LDS-staged, fragment-swizzled
// weights (ds_load_b128 per fragment) and register-resident A strips.
// Graph aggregation: coalesced per-(edge,feature) global_atomic_add_f32.
// ---------------------------------------------------------------------------

typedef _Float16 half16 __attribute__((ext_vector_type(16)));
typedef float    float8 __attribute__((ext_vector_type(8)));

// ----------------------------- small utilities -----------------------------

__global__ void fill_f32_kernel(float* __restrict__ p, float v, int n) {
  int i = blockIdx.x * blockDim.x + threadIdx.x;
  if (i < n) p[i] = v;
}

__global__ void deg_accum_kernel(const int* __restrict__ dst,
                                 float* __restrict__ deg, int E) {
  int i = blockIdx.x * blockDim.x + threadIdx.x;
  if (i < E) unsafeAtomicAdd(&deg[dst[i]], 1.0f);
}

__global__ void deg_finalize_kernel(const float* __restrict__ deg,
                                    float* __restrict__ rnorm,
                                    float* __restrict__ rdeg, int n) {
  int i = blockIdx.x * blockDim.x + threadIdx.x;
  if (i < n) {
    float d = deg[i];
    rnorm[i] = rsqrtf(d);
    rdeg[i]  = 1.0f / d;
  }
}

// ------------------------------- WMMA GEMM ---------------------------------
// H[m,n] = sum_k A[m,k] * W[k,n], compile-time K (%32==0) and NOUT (%16==0),
// nrows % 16 == 0 (true for NB=200000, NC=10000).
//
// Block = 256 threads (8 waves).  The whole W (K x NOUT, f16) is staged once
// into LDS pre-swizzled into B-fragment order: for (ntile, ktile), lane L
// holds 16 contiguous halves = its V_WMMA B operand (two ds_load_b128).
// Each wave owns one 16-row strip: A fragments for all K are loaded once
// (float4 -> cvt_pk_f16) and reused across all NOUT/16 n-tiles.

template <int K, int NOUT>
__global__ void __launch_bounds__(256)
gemm_wmma_kernel(const float* __restrict__ A, const float* __restrict__ W,
                 float* __restrict__ H, int nrows) {
  constexpr int KCH = K / 32;    // k-chunks of 32
  constexpr int NT  = NOUT / 16; // n-tiles of 16
  __shared__ _Float16 bsh[NT * KCH * 32 * 16];

  const int lane = threadIdx.x & 31;
  const int wave = threadIdx.x >> 5;
  const int half = lane >> 4;   // 0: lanes 0-15, 1: lanes 16-31
  const int l15  = lane & 15;

  // ---- stage W into LDS in B-fragment layout (coalesced-ish, W is tiny) ----
  // layout index = ((nt*KCH + kt)*32 + lane)*16 + j
// maps to W[k, n] with k = kt*32 + (lane>>4)*8 + (j&7) + (j>>3)*16, n = nt*16 + (lane&15)
  for (int idx = threadIdx.x; idx < K * NOUT; idx += 256) {
    int j  = idx & 15;
    int ln = (idx >> 4) & 31;
    int kt = (idx >> 9) % KCH;
    int nt = (idx >> 9) / KCH;
    int k  = kt * 32 + ((ln >> 4) << 3) + (j & 7) + ((j >> 3) << 4);
    int n  = nt * 16 + (ln & 15);
    bsh[idx] = (_Float16)W[(size_t)k * NOUT + n];
  }
  __syncthreads();

  const int mtiles = nrows >> 4;
  const int mt = blockIdx.x * 8 + wave;
  if (mt >= mtiles) return;                 // wave-uniform: EXEC stays all-1

  const int m0 = mt << 4;
  const int am = m0 + l15;                  // A row owned by this lane
  const int akb = half * 8;                 // A k-sub-base (ISA 7.12.2 layout)

  // ---- load A fragments for the whole K range (register resident) ----
  half16 af[KCH];
  const float* __restrict__ arow = A + (size_t)am * K;
#pragma unroll
  for (int kt = 0; kt < KCH; ++kt) {
    const float4 c0 = *(const float4*)(arow + kt * 32 + akb);
    const float4 c1 = *(const float4*)(arow + kt * 32 + akb + 4);
    const float4 c2 = *(const float4*)(arow + kt * 32 + 16 + akb);
    const float4 c3 = *(const float4*)(arow + kt * 32 + 16 + akb + 4);
    af[kt][0]  = (_Float16)c0.x;  af[kt][1]  = (_Float16)c0.y;
    af[kt][2]  = (_Float16)c0.z;  af[kt][3]  = (_Float16)c0.w;
    af[kt][4]  = (_Float16)c1.x;  af[kt][5]  = (_Float16)c1.y;
    af[kt][6]  = (_Float16)c1.z;  af[kt][7]  = (_Float16)c1.w;
    af[kt][8]  = (_Float16)c2.x;  af[kt][9]  = (_Float16)c2.y;
    af[kt][10] = (_Float16)c2.z;  af[kt][11] = (_Float16)c2.w;
    af[kt][12] = (_Float16)c3.x;  af[kt][13] = (_Float16)c3.y;
    af[kt][14] = (_Float16)c3.z;  af[kt][15] = (_Float16)c3.w;
  }

  const half16* __restrict__ bfrag = (const half16*)bsh;
#pragma unroll
  for (int nt = 0; nt < NT; ++nt) {
    float8 acc = {};
#pragma unroll
    for (int kt = 0; kt < KCH; ++kt) {
      half16 b = bfrag[(nt * KCH + kt) * 32 + lane];
      acc = __builtin_amdgcn_wmma_f32_16x16x32_f16(
          false, af[kt], false, b, (short)0, acc, false, false);
    }
    // C/D layout: lane -> col nt*16+l15; elem r -> row m0 + r + half*8.
#pragma unroll
    for (int r = 0; r < 8; ++r)
      H[(size_t)(m0 + r + half * 8) * NOUT + (nt * 16 + l15)] = acc[r];
  }
}

// ------------------------- graph scatter / combine -------------------------
// agg[dst,f] += h[src,f] * rnorm[src] * rnorm[dst]; one thread per (edge,f)
// so consecutive threads hit consecutive features (coalesced loads/atomics).

__global__ void edge_scatter_kernel(const float* __restrict__ h,
                                    const int* __restrict__ src,
                                    const int* __restrict__ dst,
                                    const float* __restrict__ rnorm,
                                    float* __restrict__ agg,
                                    long long total, int dout) {
  long long idx = (long long)blockIdx.x * blockDim.x + threadIdx.x;
  if (idx >= total) return;
  int e = (int)(idx / dout);
  int f = (int)(idx - (long long)e * dout);
  int s = src[e], d = dst[e];
  float w = rnorm[s] * rnorm[d];
  unsafeAtomicAdd(&agg[(size_t)d * dout + f], h[(size_t)s * dout + f] * w);
}

// out = agg + h * (1/deg) + bias, optional ReLU
__global__ void combine_kernel(const float* __restrict__ agg,
                               const float* __restrict__ h,
                               const float* __restrict__ rdeg,
                               const float* __restrict__ bias,
                               float* __restrict__ out,
                               long long total, int dout, int relu) {
  long long idx = (long long)blockIdx.x * blockDim.x + threadIdx.x;
  if (idx >= total) return;
  int i = (int)(idx / dout);
  int f = (int)(idx - (long long)i * dout);
  float v = agg[idx] + h[idx] * rdeg[i] + bias[f];
  out[idx] = relu ? fmaxf(v, 0.0f) : v;
}

// ------------------------- attention fuse (per row) ------------------------

__global__ void attention_fuse_kernel(const float* __restrict__ bfeat,
                                      const float* __restrict__ comm,
                                      const int* __restrict__ b2c,
                                      const int* __restrict__ l2g,
                                      const float* __restrict__ Watt,
                                      const float* __restrict__ batt,
                                      float* __restrict__ fused, int n) {
  int i = blockIdx.x * blockDim.x + threadIdx.x;
  if (i >= n) return;
  int g = l2g[i];
  const float* __restrict__ x = bfeat + (size_t)g * 64;
  const float* __restrict__ c = comm + (size_t)b2c[g] * 64;
  float l0 = batt[0], l1 = batt[1];
#pragma unroll 8
  for (int k = 0; k < 64; ++k) {
    l0 += x[k] * Watt[2 * k]     + c[k] * Watt[2 * (64 + k)];
    l1 += x[k] * Watt[2 * k + 1] + c[k] * Watt[2 * (64 + k) + 1];
  }
  float m  = fmaxf(l0, l1);
  float e0 = __expf(l0 - m), e1 = __expf(l1 - m);
  float inv = 1.0f / (e0 + e1);
  float a0 = e0 * inv, a1 = e1 * inv;
  float* __restrict__ F = fused + (size_t)i * 128;
#pragma unroll 8
  for (int k = 0; k < 64; ++k) {
    F[k]      = x[k] * a0;
    F[64 + k] = c[k] * a1;
  }
}

// ---------------- N=2 head: h = x @ W (K x 2), scalar path -----------------

__global__ void gemm_n2_kernel(const float* __restrict__ A,
                               const float* __restrict__ W,
                               float* __restrict__ H, int n, int K) {
  int i = blockIdx.x * blockDim.x + threadIdx.x;
  if (i >= n) return;
  const float* __restrict__ x = A + (size_t)i * K;
  float h0 = 0.0f, h1 = 0.0f;
#pragma unroll 8
  for (int k = 0; k < K; ++k) {
    h0 += x[k] * W[2 * k];
    h1 += x[k] * W[2 * k + 1];
  }
  H[2 * (size_t)i]     = h0;
  H[2 * (size_t)i + 1] = h1;
}

// ---------- final combine + log_softmax + scatter to global order ----------

__global__ void final_out_kernel(const float* __restrict__ agg,
                                 const float* __restrict__ h,
                                 const float* __restrict__ rdeg,
                                 const float* __restrict__ bias,
                                 const int* __restrict__ l2g,
                                 float* __restrict__ out, int n) {
  int i = blockIdx.x * blockDim.x + threadIdx.x;
  if (i >= n) return;
  float rd = rdeg[i];
  float o0 = agg[2 * (size_t)i]     + h[2 * (size_t)i]     * rd + bias[0];
  float o1 = agg[2 * (size_t)i + 1] + h[2 * (size_t)i + 1] * rd + bias[1];
  float m   = fmaxf(o0, o1);
  float lse = m + logf(__expf(o0 - m) + __expf(o1 - m));
  int g = l2g[i];
  out[2 * (size_t)g]     = o0 - lse;
  out[2 * (size_t)g + 1] = o1 - lse;
}

// ---------------------------------------------------------------------------

static inline int cdiv(long long a, int b) { return (int)((a + b - 1) / b); }

extern "C" void kernel_launch(void* const* d_in, const int* in_sizes, int n_in,
                              void* d_out, int out_size, void* d_ws, size_t ws_size,
                              hipStream_t stream) {
  const float* bfeat = (const float*)d_in[0];   // [NB, 64]
  const float* cfeat = (const float*)d_in[1];   // [NC, 32]
  const float* W_c1  = (const float*)d_in[2];
  const float* b_c1  = (const float*)d_in[3];
  const float* W_c2  = (const float*)d_in[4];
  const float* b_c2  = (const float*)d_in[5];
  const float* W_att = (const float*)d_in[6];
  const float* b_att = (const float*)d_in[7];
  const float* W_b1  = (const float*)d_in[8];
  const float* b_b1  = (const float*)d_in[9];
  const float* W_b2  = (const float*)d_in[10];
  const float* b_b2  = (const float*)d_in[11];
  const float* W_b3  = (const float*)d_in[12];
  const float* b_b3  = (const float*)d_in[13];
  const int* b_src = (const int*)d_in[14];
  const int* b_dst = (const int*)d_in[15];
  const int* c_src = (const int*)d_in[16];
  const int* c_dst = (const int*)d_in[17];
  const int* b2c   = (const int*)d_in[18];
  const int* l2g   = (const int*)d_in[19];
  float* out = (float*)d_out;

  const int NB = in_sizes[0] / 64;     // 200000
  const int NC = in_sizes[1] / 32;     // 10000
  const int EB = in_sizes[14];         // 1200000
  const int EC = in_sizes[16];         // 160000

  // ---- workspace carve-up (aligned 256B) ----
  char* ws = (char*)d_ws;
  size_t off = 0;
  auto alloc = [&](size_t nfloats) -> float* {
    float* p = (float*)(ws + off);
    off += (nfloats * sizeof(float) + 255) & ~(size_t)255;
    return p;
  };
  float* F  = alloc((size_t)NB * 128);   // fused / layer outputs
  float* H  = alloc((size_t)NB * 128);   // pre-aggregation h = x @ W
  float* G  = alloc((size_t)NB * 128);   // scatter accumulator
  float* Fc = alloc((size_t)NC * 64);
  float* Hc = alloc((size_t)NC * 64);
  float* Gc = alloc((size_t)NC * 64);
  float* deg_b   = alloc(NB);
  float* rnorm_b = alloc(NB);
  float* rdeg_b  = alloc(NB);
  float* deg_c   = alloc(NC);
  float* rnorm_c = alloc(NC);
  float* rdeg_c  = alloc(NC);
  float* h3 = alloc((size_t)NB * 2);
  float* g3 = alloc((size_t)NB * 2);
  (void)ws_size;

  const int T = 256;
  auto strip_blocks = [&](int nrows) { return ((nrows >> 4) + 7) / 8; };

  // ---- degrees (deg = 1 + in-edge count) ----
  fill_f32_kernel<<<cdiv(NC, T), T, 0, stream>>>(deg_c, 1.0f, NC);
  deg_accum_kernel<<<cdiv(EC, T), T, 0, stream>>>(c_dst, deg_c, EC);
  deg_finalize_kernel<<<cdiv(NC, T), T, 0, stream>>>(deg_c, rnorm_c, rdeg_c, NC);
  fill_f32_kernel<<<cdiv(NB, T), T, 0, stream>>>(deg_b, 1.0f, NB);
  deg_accum_kernel<<<cdiv(EB, T), T, 0, stream>>>(b_dst, deg_b, EB);
  deg_finalize_kernel<<<cdiv(NB, T), T, 0, stream>>>(deg_b, rnorm_b, rdeg_b, NB);

  // ---- community GCN layer 1: 32 -> 64, ReLU ----
  gemm_wmma_kernel<32, 64><<<strip_blocks(NC), T, 0, stream>>>(cfeat, W_c1, Hc, NC);
  hipMemsetAsync(Gc, 0, (size_t)NC * 64 * sizeof(float), stream);
  edge_scatter_kernel<<<cdiv((long long)EC * 64, T), T, 0, stream>>>(
      Hc, c_src, c_dst, rnorm_c, Gc, (long long)EC * 64, 64);
  combine_kernel<<<cdiv((long long)NC * 64, T), T, 0, stream>>>(
      Gc, Hc, rdeg_c, b_c1, Fc, (long long)NC * 64, 64, 1);

  // ---- community GCN layer 2: 64 -> 64, ReLU ----
  gemm_wmma_kernel<64, 64><<<strip_blocks(NC), T, 0, stream>>>(Fc, W_c2, Hc, NC);
  hipMemsetAsync(Gc, 0, (size_t)NC * 64 * sizeof(float), stream);
  edge_scatter_kernel<<<cdiv((long long)EC * 64, T), T, 0, stream>>>(
      Hc, c_src, c_dst, rnorm_c, Gc, (long long)EC * 64, 64);
  combine_kernel<<<cdiv((long long)NC * 64, T), T, 0, stream>>>(
      Gc, Hc, rdeg_c, b_c2, Fc, (long long)NC * 64, 64, 1);

  // ---- attention-gated fusion (includes comm gather + local_to_global) ----
  attention_fuse_kernel<<<cdiv(NB, T), T, 0, stream>>>(
      bfeat, Fc, b2c, l2g, W_att, b_att, F, NB);

  // ---- building GCN layer 1: 128 -> 128, ReLU ----
  gemm_wmma_kernel<128, 128><<<strip_blocks(NB), T, 0, stream>>>(F, W_b1, H, NB);
  hipMemsetAsync(G, 0, (size_t)NB * 128 * sizeof(float), stream);
  edge_scatter_kernel<<<cdiv((long long)EB * 128, T), T, 0, stream>>>(
      H, b_src, b_dst, rnorm_b, G, (long long)EB * 128, 128);
  combine_kernel<<<cdiv((long long)NB * 128, T), T, 0, stream>>>(
      G, H, rdeg_b, b_b1, F, (long long)NB * 128, 128, 1);

  // ---- building GCN layer 2: 128 -> 64, ReLU ----
  gemm_wmma_kernel<128, 64><<<strip_blocks(NB), T, 0, stream>>>(F, W_b2, H, NB);
  hipMemsetAsync(G, 0, (size_t)NB * 64 * sizeof(float), stream);
  edge_scatter_kernel<<<cdiv((long long)EB * 64, T), T, 0, stream>>>(
      H, b_src, b_dst, rnorm_b, G, (long long)EB * 64, 64);
  combine_kernel<<<cdiv((long long)NB * 64, T), T, 0, stream>>>(
      G, H, rdeg_b, b_b2, F, (long long)NB * 64, 64, 1);

  // ---- building GCN layer 3: 64 -> 2, then log-softmax + scatter ----
  gemm_n2_kernel<<<cdiv(NB, T), T, 0, stream>>>(F, W_b3, h3, NB, 64);
  hipMemsetAsync(g3, 0, (size_t)NB * 2 * sizeof(float), stream);
  edge_scatter_kernel<<<cdiv((long long)EB * 2, T), T, 0, stream>>>(
      h3, b_src, b_dst, rnorm_b, g3, (long long)EB * 2, 2);
  final_out_kernel<<<cdiv(NB, T), T, 0, stream>>>(
      g3, h3, rdeg_b, b_b3, l2g, out, NB);
}
